// DSTMamba_17282948399326
// MI455X (gfx1250) — matrix operand order repaired and checked by reading, compiled, not run on hardware
//
#include <hip/hip_runtime.h>
#include <hip/hip_bf16.h>
#include <math.h>

// ---------------- model dimensions ----------------
#define BB    8
#define LHIST 512
#define CC    862
#define EE    448          // EMBED = D_MODEL - NODE_DIM
#define DMOD  512
#define NDIM  64           // NODE_DIM
#define RNK   32           // RANK
#define DIN   1024         // D_INNER
#define DSTT  16           // D_STATE
#define DTRK  32           // DT_RANK
#define DFF   2048
#define LFUT  96
#define MTOK  (BB*CC)      // 6896 tokens (multiple of 16)

typedef __attribute__((ext_vector_type(8)))  _Float16 v8h;
typedef __attribute__((ext_vector_type(16))) _Float16 v16h;
typedef __attribute__((ext_vector_type(8)))  float    v8f;

// ================= kernels =================

__global__ void k_cvt_f16(const float* __restrict__ s, _Float16* __restrict__ d, long n) {
    long i = (long)blockIdx.x * blockDim.x + threadIdx.x;
    if (i < n) d[i] = (_Float16)s[i];
}

__global__ void k_fill0(float* __restrict__ p, long n) {
    long i = (long)blockIdx.x * blockDim.x + threadIdx.x;
    if (i < n) p[i] = 0.0f;
}

// per-(b,c) mean / stdev over L
__global__ void k_revin_stats(const float* __restrict__ x, float* __restrict__ mean,
                              float* __restrict__ stdev) {
    int r = blockIdx.x * blockDim.x + threadIdx.x;
    if (r >= BB * CC) return;
    int b = r / CC, c = r % CC;
    float s = 0.f, s2 = 0.f;
    for (int l = 0; l < LHIST; ++l) {
        float v = x[((long)b * LHIST + l) * CC + c];
        s += v; s2 += v * v;
    }
    float m = s / LHIST;
    float var = s2 / LHIST - m * m;
    mean[r] = m;
    stdev[r] = sqrtf(var + 1e-5f);
}

__global__ void k_revin_apply(const float* __restrict__ x, const float* __restrict__ mean,
                              const float* __restrict__ stdev, const float* __restrict__ rw,
                              const float* __restrict__ rb, float* __restrict__ xn) {
    long i = (long)blockIdx.x * blockDim.x + threadIdx.x;
    if (i >= (long)BB * LHIST * CC) return;
    int c = (int)(i % CC);
    int b = (int)(i / ((long)LHIST * CC));
    float v = (x[i] - mean[b * CC + c]) / stdev[b * CC + c];
    xn[i] = v * rw[c] + rb[c];
}

// moving average, window 25, edge-replicated (thread per (b,c) row)
__global__ void k_movavg(const float* __restrict__ in, float* __restrict__ ma, int Lc) {
    int r = blockIdx.x * blockDim.x + threadIdx.x;
    if (r >= BB * CC) return;
    int b = r / CC, c = r % CC;
    for (int t = 0; t < Lc; ++t) {
        float s = 0.f;
        for (int j = -12; j <= 12; ++j) {
            int tt = t + j; tt = tt < 0 ? 0 : (tt >= Lc ? Lc - 1 : tt);
            s += in[((long)b * Lc + tt) * CC + c];
        }
        ma[((long)b * Lc + t) * CC + c] = s * (1.0f / 25.0f);
    }
}

// x_sea transposed to (b,c,l) for the embedding GEMM
__global__ void k_seaT(const float* __restrict__ xn, const float* __restrict__ ma,
                       float* __restrict__ seaT) {
    long i = (long)blockIdx.x * blockDim.x + threadIdx.x;
    if (i >= (long)BB * LHIST * CC) return;
    int c = (int)(i % CC);
    long rem = i / CC;
    int l = (int)(rem % LHIST);
    int b = (int)(rem / LHIST);
    seaT[((long)b * CC + c) * LHIST + l] = xn[i] - ma[i];
}

// -------- WMMA GEMM, double-buffered async global->LDS pipeline --------
// out[M,N] = act(A[M,K(lda)] * W[N,K]^T + bias); act: 0=none 1=relu 2=softplus
// grid: ((N+63)/64, M/16), block 128 (4 waves, each one 16x16 N-subtile).
// Each 64-wide K stage (A tile 16x64, W tile 64x64) is staged with
// GLOBAL_LOAD_ASYNC_TO_LDS_B128 (exactly 5 async instrs per wave per stage);
// stage s+1 is prefetched into the other buffer and `s_wait_asynccnt 5`
// overlaps its DMA with the WMMAs of stage s.
__global__ void k_wmma_gemm(const _Float16* __restrict__ A, long lda,
                            const _Float16* __restrict__ W,
                            const float* __restrict__ bias,
                            float* __restrict__ Cout, long ldc,
                            int M, int N, int K, int act, int accum) {
    __shared__ __align__(16) _Float16 At[2][16 * 64];   // row stride 64 halves
    __shared__ __align__(16) _Float16 Wt[2][64 * 64];

    int tid  = threadIdx.x;
    int lane = tid & 31;
    int wv   = tid >> 5;
    int n0base = blockIdx.x * 64;
    int n0 = n0base + wv * 16;
    int m0 = blockIdx.y * 16;
    bool active = (n0 < N);          // no early return: all waves reach barriers

    int kgrp  = (lane >> 4) * 8;     // ISA 16-bit A fragment K-group
    int kbase = (lane >> 4) * 16;    // unused; B is loaded with kgrp, same k order as A

    // async-copy slot for the A tile: one B128 (8 halves) per thread
    int arow = tid >> 3;             // 0..15
    int aseg = tid & 7;              // 0..7
    const _Float16* a_g0 = A + (long)(m0 + arow) * lda + aseg * 8;

    auto stage_load = [&](int k0, int buf) {
        // A tile: 1 async instr per wave
        if (k0 + aseg * 8 < K) {
            unsigned a_lds = (unsigned)(unsigned long long)(const void*)
                             &At[buf][arow * 64 + aseg * 8];
            unsigned long long ga = (unsigned long long)(a_g0 + k0);
            asm volatile("global_load_async_to_lds_b128 %0, %1, off"
                         :: "v"(a_lds), "v"(ga) : "memory");
        }
        // W tile: 4 async instrs per wave
#pragma unroll
        for (int j = 0; j < 4; ++j) {
            int slot = tid + 128 * j;
            int row = slot >> 3, seg = slot & 7;
            int wr = n0base + row; if (wr >= N) wr = N - 1;   // clamp; masked at store
            if (k0 + seg * 8 < K) {
                unsigned w_lds = (unsigned)(unsigned long long)(const void*)
                                 &Wt[buf][row * 64 + seg * 8];
                unsigned long long gw =
                    (unsigned long long)(W + (long)wr * K + k0 + seg * 8);
                asm volatile("global_load_async_to_lds_b128 %0, %1, off"
                             :: "v"(w_lds), "v"(gw) : "memory");
            }
        }
    };

    v8f acc;
#pragma unroll
    for (int i = 0; i < 8; ++i) acc[i] = 0.0f;

    stage_load(0, 0);                                    // prologue

    int nstages = (K + 63) / 64;
    for (int s = 0; s < nstages; ++s) {
        int k0 = s * 64;
        int buf = s & 1;
        if (k0 + 64 < K) {                               // prefetch next stage
            stage_load(k0 + 64, buf ^ 1);
            // current stage complete; next stage's 5 per-wave loads may remain
            asm volatile("s_wait_asynccnt 0x5" ::: "memory");
        } else {
            asm volatile("s_wait_asynccnt 0x0" ::: "memory");
        }
        __syncthreads();                                 // stage visible to all waves

        // ---- consume: up to two 16x16x32 WMMAs from LDS ----
#pragma unroll
        for (int kk = 0; kk < 64; kk += 32) {
            if (k0 + kk < K) {
                const _Float16* ar = &At[buf][(lane & 15) * 64 + kk];
                v8h alo = *(const v8h*)(ar + kgrp);
                v8h ahi = *(const v8h*)(ar + 16 + kgrp);
                v16h av = __builtin_shufflevector(alo, ahi,
                          0,1,2,3,4,5,6,7,8,9,10,11,12,13,14,15);
                const _Float16* br = &Wt[buf][(wv * 16 + (lane & 15)) * 64 + kk];
                v8h blo = *(const v8h*)(br + kgrp);
                v8h bhi = *(const v8h*)(br + 16 + kgrp);
                v16h bv = __builtin_shufflevector(blo, bhi,
                          0,1,2,3,4,5,6,7,8,9,10,11,12,13,14,15);
                acc = __builtin_amdgcn_wmma_f32_16x16x32_f16(false, av, false, bv,
                                                             (short)0, acc, false, false);
            }
        }
        __syncthreads();                 // done reading buf before it is re-staged
    }

    if (active) {
        int ncol = n0 + (lane & 15);
        float bval = bias ? bias[ncol] : 0.0f;
#pragma unroll
        for (int i = 0; i < 8; ++i) {
            int m = m0 + i + 8 * (lane >> 4);
            float v = acc[i] + bval;
            if (act == 1) v = v > 0.f ? v : 0.f;
            else if (act == 2) v = __logf(1.0f + __expf(v));   // softplus, fast path
            float* dst = Cout + (long)m * ldc + ncol;
            if (accum) *dst += v; else *dst = v;
        }
    }
}

__global__ void k_concat_emb(const float* __restrict__ xemb, float* __restrict__ h) {
    long i = (long)blockIdx.x * blockDim.x + threadIdx.x;
    if (i >= (long)MTOK * EE) return;
    long m = i / EE; int e = (int)(i % EE);
    h[m * DMOD + e] = xemb[i];
}

// adapt[b,c,d] = sum_e xemb[b,c,e] * adapter_act[c,e,d]  -> h[...,448+d]
__global__ void k_adapt(const float* __restrict__ xemb, const float* __restrict__ aact,
                        float* __restrict__ h) {
    long i = (long)blockIdx.x * blockDim.x + threadIdx.x;
    if (i >= (long)MTOK * NDIM) return;
    long m = i / NDIM; int d = (int)(i % NDIM);
    int c = (int)(m % CC);
    const float* xr = xemb + m * EE;
    const float* ar = aact + (long)c * EE * NDIM + d;
    float s = 0.f;
    for (int e = 0; e < EE; ++e) s += xr[e] * ar[(long)e * NDIM];
    h[m * DMOD + EE + d] = s;
}

// causal depthwise conv (D_CONV=4) + silu; sequence dim is the channel-token axis
__global__ void k_conv_silu(const float* __restrict__ xz, const float* __restrict__ cw,
                            const float* __restrict__ cb, float* __restrict__ xc, int rev) {
    long i = (long)blockIdx.x * blockDim.x + threadIdx.x;
    if (i >= (long)MTOK * DIN) return;
    long m = i / DIN; int d = (int)(i % DIN);
    int b = (int)(m / CC), t = (int)(m % CC);
    float s = cb[d];
#pragma unroll
    for (int k = 0; k < 4; ++k) {
        int tt = rev ? (t + 3 - k) : (t + k - 3);
        if (tt >= 0 && tt < CC)
            s += xz[((long)b * CC + tt) * (2 * DIN) + d] * cw[d * 4 + k];
    }
    xc[i] = s / (1.0f + __expf(-s)) ;   // silu
}

// selective scan, state in registers; gating with silu(z) folded in
__global__ void k_scan(const float* __restrict__ dt, const float* __restrict__ xc,
                       const float* __restrict__ dbl, const float* __restrict__ xz,
                       const float* __restrict__ A_log, const float* __restrict__ Dp,
                       float* __restrict__ y, int rev) {
    int tid = blockIdx.x * blockDim.x + threadIdx.x;
    if (tid >= BB * DIN) return;
    int b = tid / DIN, d = tid % DIN;
    float Ac[DSTT], hs[DSTT];
#pragma unroll
    for (int s = 0; s < DSTT; ++s) { Ac[s] = -__expf(A_log[d * DSTT + s]); hs[s] = 0.f; }
    float Dd = Dp[d];
    for (int step = 0; step < CC; ++step) {
        int t = rev ? (CC - 1 - step) : step;
        long m = (long)b * CC + t;
        float dtv = dt[m * DIN + d];
        float xv  = xc[m * DIN + d];
        float dx  = dtv * xv;
        float acc = 0.f;
#pragma unroll
        for (int s = 0; s < DSTT; ++s) {
            float e = __expf(dtv * Ac[s]);
            hs[s] = e * hs[s] + dx * dbl[m * 64 + DTRK + s];
            acc  += hs[s] * dbl[m * 64 + DTRK + DSTT + s];
        }
        float yv = acc + xv * Dd;
        float z  = xz[m * (2 * DIN) + DIN + d];
        yv *= z / (1.0f + __expf(-z));
        y[m * DIN + d] = yv;
    }
}

// out = layernorm(h + add) over DMOD; add may be null; in-place safe
__global__ void k_add_ln(const float* __restrict__ h, const float* __restrict__ add,
                         const float* __restrict__ g, const float* __restrict__ bt,
                         float* __restrict__ out) {
    int m = blockIdx.x * blockDim.x + threadIdx.x;
    if (m >= MTOK) return;
    long base = (long)m * DMOD;
    float mean = 0.f;
    for (int i = 0; i < DMOD; ++i) mean += h[base + i] + (add ? add[base + i] : 0.f);
    mean *= (1.0f / DMOD);
    float var = 0.f;
    for (int i = 0; i < DMOD; ++i) {
        float v = h[base + i] + (add ? add[base + i] : 0.f) - mean;
        var += v * v;
    }
    var *= (1.0f / DMOD);
    float inv = rsqrtf(var + 1e-5f);
    for (int i = 0; i < DMOD; ++i) {
        float v = h[base + i] + (add ? add[base + i] : 0.f);
        out[base + i] = (v - mean) * inv * g[i] + bt[i];
    }
}

__global__ void k_downsample(const float* __restrict__ in, float* __restrict__ out, int Lin) {
    int Lout = Lin / 2;
    long i = (long)blockIdx.x * blockDim.x + threadIdx.x;
    if (i >= (long)BB * Lout * CC) return;
    int c = (int)(i % CC);
    long rem = i / CC;
    int l = (int)(rem % Lout), b = (int)(rem / Lout);
    out[i] = 0.5f * (in[((long)b * Lin + 2 * l) * CC + c] +
                     in[((long)b * Lin + 2 * l + 1) * CC + c]);
}

// out[b,m,c] = (accum? out : 0) + (addsrc? addsrc : 0) + act(sum_l in[b,l,c]*w[m,l]+bias[m])
// act: 0=none 3=gelu(exact)
__global__ void k_vecmap(const float* __restrict__ in, int Lin,
                         const float* __restrict__ w, const float* __restrict__ bias,
                         const float* __restrict__ addsrc,
                         float* __restrict__ out, int Lout, int act, int accum) {
    long i = (long)blockIdx.x * blockDim.x + threadIdx.x;
    if (i >= (long)BB * Lout * CC) return;
    int c = (int)(i % CC);
    long rem = i / CC;
    int m = (int)(rem % Lout), b = (int)(rem / Lout);
    float s = bias ? bias[m] : 0.f;
    const float* wr = w + (long)m * Lin;
    for (int l = 0; l < Lin; ++l) s += in[((long)b * Lin + l) * CC + c] * wr[l];
    if (act == 3) s = 0.5f * s * (1.0f + erff(s * 0.70710678f));
    if (addsrc) s += addsrc[i];
    if (accum) out[i] += s; else out[i] = s;
}

__global__ void k_final(const float* __restrict__ sea_bcf, const float* __restrict__ tre,
                        const float* __restrict__ tre_w, const float* __restrict__ rb,
                        const float* __restrict__ rw, const float* __restrict__ stdev,
                        const float* __restrict__ mean, float* __restrict__ outp) {
    long i = (long)blockIdx.x * blockDim.x + threadIdx.x;
    if (i >= (long)BB * LFUT * CC) return;
    int c = (int)(i % CC);
    long rem = i / CC;
    int f = (int)(rem % LFUT), b = (int)(rem / LFUT);
    float comb = sea_bcf[((long)b * CC + c) * LFUT + f] + tre_w[c] * tre[i];
    float v = (comb - rb[c]) / (rw[c] + 1e-10f);
    outp[i] = v * stdev[b * CC + c] + mean[b * CC + c];
}

// ================= host side =================

static inline dim3 g1(long n, int bs) { return dim3((unsigned)((n + bs - 1) / bs)); }

static void cvt(hipStream_t st, const float* s, _Float16* d, long n) {
    k_cvt_f16<<<g1(n, 256), 256, 0, st>>>(s, d, n);
}

static void gemm(hipStream_t st, const _Float16* A, long lda, const _Float16* W,
                 const float* bias, float* C, long ldc,
                 int M, int N, int K, int act, int accum) {
    dim3 grid((N + 63) / 64, M / 16);
    k_wmma_gemm<<<grid, 128, 0, st>>>(A, lda, W, bias, C, ldc, M, N, K, act, accum);
}

struct SSMp {
    const float *in_proj, *conv_w, *conv_b, *x_proj, *dt_w, *dt_b, *A_log, *D, *out_proj;
};
struct Layerp {
    SSMp ssm, ssm_r;
    const float *w1, *b1, *w2, *b2, *n1g, *n1b, *n2g, *n2b;
};

extern "C" void kernel_launch(void* const* d_in, const int* in_sizes, int n_in,
                              void* d_out, int out_size, void* d_ws, size_t ws_size,
                              hipStream_t stream) {
    (void)in_sizes; (void)n_in; (void)out_size; (void)ws_size;
    int idx = 0;
    auto nxt = [&]() { return (const float*)d_in[idx++]; };

    const float* hist    = nxt();
    const float* revin_w = nxt();
    const float* revin_b = nxt();
    const float* emb_w   = nxt();
    const float* emb_b   = nxt();
    const float* adapterP= nxt();
    const float* lora_w  = nxt();

    Layerp L[2];
    for (int l = 0; l < 2; ++l) {
        SSMp* ss[2] = { &L[l].ssm, &L[l].ssm_r };
        for (int k = 0; k < 2; ++k) {
            SSMp& s = *ss[k];
            s.in_proj = nxt(); s.conv_w = nxt(); s.conv_b = nxt();
            s.x_proj = nxt();  s.dt_w = nxt();   s.dt_b = nxt();
            s.A_log = nxt();   s.D = nxt();      s.out_proj = nxt();
        }
        L[l].w1 = nxt(); L[l].b1 = nxt(); L[l].w2 = nxt(); L[l].b2 = nxt();
        L[l].n1g = nxt(); L[l].n1b = nxt(); L[l].n2g = nxt(); L[l].n2b = nxt();
    }
    const float* norm_g = nxt();
    const float* norm_b = nxt();
    const float* proj_w = nxt();
    const float* proj_b = nxt();
    const float *up_w1[3], *up_b1[3], *up_w2[3], *up_b2[3];
    for (int i = 0; i < 3; ++i) { up_w1[i] = nxt(); up_b1[i] = nxt(); up_w2[i] = nxt(); up_b2[i] = nxt(); }
    const float *map_w[4], *map_b[4];
    for (int i = 0; i < 4; ++i) { map_w[i] = nxt(); map_b[i] = nxt(); }
    const float* tre_w = nxt();

    // ---- workspace bump allocator ----
    size_t off = 0;
    auto allocf = [&](long n) { float* p = (float*)((char*)d_ws + off);
        off += (((size_t)n * 4) + 255) & ~(size_t)255; return p; };
    auto alloch = [&](long n) { _Float16* p = (_Float16*)((char*)d_ws + off);
        off += (((size_t)n * 2) + 255) & ~(size_t)255; return p; };

    const long NLC = (long)BB * LHIST * CC;
    float* meanb  = allocf(MTOK);
    float* stdevb = allocf(MTOK);
    float* xn     = allocf(NLC);
    float* tr0    = allocf(NLC);                     // ma of xn (also sea trend)
    float* seaT   = allocf(NLC);
    _Float16* seaT16 = alloch(NLC);
    float* ms1 = allocf((long)BB * 256 * CC);
    float* ms2 = allocf((long)BB * 128 * CC);
    float* ms3 = allocf((long)BB * 64 * CC);
    float* tr1 = allocf((long)BB * 256 * CC);
    float* tr2 = allocf((long)BB * 128 * CC);
    float* tr3 = allocf((long)BB * 64 * CC);
    float* xemb = allocf((long)MTOK * EE);
    _Float16* adapter16 = alloch((long)CC * EE * RNK);
    _Float16* lora16    = alloch(NDIM * RNK);
    float* aact = allocf((long)CC * EE * NDIM);
    float* h    = allocf((long)MTOK * DMOD);
    _Float16* h16 = alloch((long)MTOK * DMOD);
    float* xz  = allocf((long)MTOK * 2 * DIN);
    float* xc  = allocf((long)MTOK * DIN);
    _Float16* xc16 = alloch((long)MTOK * DIN);
    float* dbl = allocf((long)MTOK * 64);
    _Float16* dbl16 = alloch((long)MTOK * 64);
    float* dtb = allocf((long)MTOK * DIN);
    float* ybuf = allocf((long)MTOK * DIN);
    _Float16* y16 = alloch((long)MTOK * DIN);
    float* mbuf = allocf((long)MTOK * DMOD);
    float* ffn1 = allocf((long)MTOK * DFF);
    _Float16* ffn1_16 = alloch((long)MTOK * DFF);
    float* ffn2 = allocf((long)MTOK * DMOD);
    _Float16* enc16 = alloch((long)MTOK * DMOD);
    float* sea_bcf = allocf((long)MTOK * LFUT);
    float* tre = allocf((long)BB * LFUT * CC);
    float* tmpA = allocf(NLC);
    float* out128 = allocf((long)BB * 128 * CC);
    float* out256 = allocf((long)BB * 256 * CC);
    float* out512 = allocf(NLC);

    _Float16* emb16  = alloch((long)EE * LHIST);
    _Float16* projw16 = alloch((long)LFUT * DMOD);
    _Float16 *w_in16[4], *w_xp16[4], *w_dt16[4], *w_out16[4];
    _Float16 *w_ff1[2], *w_ff2[2];
    for (int i = 0; i < 4; ++i) {
        w_in16[i]  = alloch((long)2 * DIN * DMOD);
        w_xp16[i]  = alloch((long)64 * DIN);
        w_dt16[i]  = alloch((long)DIN * DTRK);
        w_out16[i] = alloch((long)DMOD * DIN);
    }
    for (int i = 0; i < 2; ++i) {
        w_ff1[i] = alloch((long)DFF * DMOD);
        w_ff2[i] = alloch((long)DMOD * DFF);
    }

    // ---- weight conversions to f16 ----
    cvt(stream, emb_w, emb16, (long)EE * LHIST);
    cvt(stream, proj_w, projw16, (long)LFUT * DMOD);
    cvt(stream, adapterP, adapter16, (long)CC * EE * RNK);
    cvt(stream, lora_w, lora16, NDIM * RNK);
    for (int l = 0; l < 2; ++l) {
        const SSMp* ss[2] = { &L[l].ssm, &L[l].ssm_r };
        for (int k = 0; k < 2; ++k) {
            int q = l * 2 + k;
            cvt(stream, ss[k]->in_proj,  w_in16[q],  (long)2 * DIN * DMOD);
            cvt(stream, ss[k]->x_proj,   w_xp16[q],  (long)64 * DIN);
            cvt(stream, ss[k]->dt_w,     w_dt16[q],  (long)DIN * DTRK);
            cvt(stream, ss[k]->out_proj, w_out16[q], (long)DMOD * DIN);
        }
        cvt(stream, L[l].w1, w_ff1[l], (long)DFF * DMOD);
        cvt(stream, L[l].w2, w_ff2[l], (long)DMOD * DFF);
    }

    // ---- prologue: RevIN, decomposition, embedding, adapter ----
    k_revin_stats<<<g1(MTOK, 256), 256, 0, stream>>>(hist, meanb, stdevb);
    k_revin_apply<<<g1(NLC, 256), 256, 0, stream>>>(hist, meanb, stdevb, revin_w, revin_b, xn);
    k_movavg<<<g1(MTOK, 128), 128, 0, stream>>>(xn, tr0, LHIST);
    k_seaT<<<g1(NLC, 256), 256, 0, stream>>>(xn, tr0, seaT);
    cvt(stream, seaT, seaT16, NLC);
    gemm(stream, seaT16, LHIST, emb16, emb_b, xemb, EE, MTOK, EE, LHIST, 0, 0);
    gemm(stream, adapter16, RNK, lora16, nullptr, aact, NDIM, CC * EE, NDIM, RNK, 1, 0);
    k_concat_emb<<<g1((long)MTOK * EE, 256), 256, 0, stream>>>(xemb, h);
    k_adapt<<<g1((long)MTOK * NDIM, 256), 256, 0, stream>>>(xemb, aact, h);

    // ---- encoder layers ----
    for (int l = 0; l < 2; ++l) {
        cvt(stream, h, h16, (long)MTOK * DMOD);
        for (int r = 0; r < 2; ++r) {            // r=0 forward, r=1 reversed
            int q = l * 2 + r;
            const SSMp& s = r ? L[l].ssm_r : L[l].ssm;
            gemm(stream, h16, DMOD, w_in16[q], nullptr, xz, 2 * DIN, MTOK, 2 * DIN, DMOD, 0, 0);
            k_conv_silu<<<g1((long)MTOK * DIN, 256), 256, 0, stream>>>(xz, s.conv_w, s.conv_b, xc, r);
            cvt(stream, xc, xc16, (long)MTOK * DIN);
            gemm(stream, xc16, DIN, w_xp16[q], nullptr, dbl, 64, MTOK, 64, DIN, 0, 0);
            cvt(stream, dbl, dbl16, (long)MTOK * 64);
            gemm(stream, dbl16, 64, w_dt16[q], s.dt_b, dtb, DIN, MTOK, DIN, DTRK, 2, 0);
            k_scan<<<g1(BB * DIN, 256), 256, 0, stream>>>(dtb, xc, dbl, xz, s.A_log, s.D, ybuf, r);
            cvt(stream, ybuf, y16, (long)MTOK * DIN);
            gemm(stream, y16, DIN, w_out16[q], nullptr, mbuf, DMOD, MTOK, DMOD, DIN, 0, r);
        }
        k_add_ln<<<g1(MTOK, 128), 128, 0, stream>>>(h, mbuf, L[l].n1g, L[l].n1b, h);
        cvt(stream, h, h16, (long)MTOK * DMOD);
        gemm(stream, h16, DMOD, w_ff1[l], L[l].b1, ffn1, DFF, MTOK, DFF, DMOD, 1, 0);
        cvt(stream, ffn1, ffn1_16, (long)MTOK * DFF);
        gemm(stream, ffn1_16, DFF, w_ff2[l], L[l].b2, ffn2, DMOD, MTOK, DMOD, DFF, 0, 0);
        k_add_ln<<<g1(MTOK, 128), 128, 0, stream>>>(h, ffn2, L[l].n2g, L[l].n2b, h);
    }

    // ---- final norm + seasonal projection ----
    k_add_ln<<<g1(MTOK, 128), 128, 0, stream>>>(h, nullptr, norm_g, norm_b, h);
    cvt(stream, h, enc16, (long)MTOK * DMOD);
    gemm(stream, enc16, DMOD, projw16, proj_b, sea_bcf, LFUT, MTOK, LFUT, DMOD, 0, 0);

    // ---- multi-scale trend path ----
    k_downsample<<<g1((long)BB * 256 * CC, 256), 256, 0, stream>>>(xn, ms1, 512);
    k_downsample<<<g1((long)BB * 128 * CC, 256), 256, 0, stream>>>(ms1, ms2, 256);
    k_downsample<<<g1((long)BB * 64 * CC, 256), 256, 0, stream>>>(ms2, ms3, 128);
    k_movavg<<<g1(MTOK, 128), 128, 0, stream>>>(ms1, tr1, 256);
    k_movavg<<<g1(MTOK, 128), 128, 0, stream>>>(ms2, tr2, 128);
    k_movavg<<<g1(MTOK, 128), 128, 0, stream>>>(ms3, tr3, 64);

    // fusion: tr3(64) -> +tr2 => out128 -> +tr1 => out256 -> +tr0 => out512
    k_vecmap<<<g1((long)BB*128*CC,256),256,0,stream>>>(tr3, 64, up_w1[0], up_b1[0], nullptr, tmpA, 128, 3, 0);
    k_vecmap<<<g1((long)BB*128*CC,256),256,0,stream>>>(tmpA, 128, up_w2[0], up_b2[0], tr2, out128, 128, 0, 0);
    k_vecmap<<<g1((long)BB*256*CC,256),256,0,stream>>>(out128, 128, up_w1[1], up_b1[1], nullptr, tmpA, 256, 3, 0);
    k_vecmap<<<g1((long)BB*256*CC,256),256,0,stream>>>(tmpA, 256, up_w2[1], up_b2[1], tr1, out256, 256, 0, 0);
    k_vecmap<<<g1((long)BB*512*CC,256),256,0,stream>>>(out256, 256, up_w1[2], up_b1[2], nullptr, tmpA, 512, 3, 0);
    k_vecmap<<<g1((long)BB*512*CC,256),256,0,stream>>>(tmpA, 512, up_w2[2], up_b2[2], tr0, out512, 512, 0, 0);

    // trend -> future maps (accumulated)
    k_fill0<<<g1((long)BB*LFUT*CC,256),256,0,stream>>>(tre, (long)BB*LFUT*CC);
    k_vecmap<<<g1((long)BB*LFUT*CC,256),256,0,stream>>>(out512, 512, map_w[0], map_b[0], nullptr, tre, LFUT, 0, 1);
    k_vecmap<<<g1((long)BB*LFUT*CC,256),256,0,stream>>>(out256, 256, map_w[1], map_b[1], nullptr, tre, LFUT, 0, 1);
    k_vecmap<<<g1((long)BB*LFUT*CC,256),256,0,stream>>>(out128, 128, map_w[2], map_b[2], nullptr, tre, LFUT, 0, 1);
    k_vecmap<<<g1((long)BB*LFUT*CC,256),256,0,stream>>>(tr3,    64,  map_w[3], map_b[3], nullptr, tre, LFUT, 0, 1);

    // ---- combine + inverse RevIN ----
    k_final<<<g1((long)BB*LFUT*CC,256),256,0,stream>>>(sea_bcf, tre, tre_w, revin_b, revin_w,
                                                       stdevb, meanb, (float*)d_out);
}
